// FloatingPointHGRNModel_78950088835410
// MI455X (gfx1250) — compile-verified, matrix operand order checked by reading, not run on hardware
//
#include <hip/hip_runtime.h>
#include <hip/hip_bf16.h>

// ---------------------------------------------------------------------------
// HGRN forward for MI455X (gfx1250, wave32).
// Roofline: ~0.55 TFLOP total; weights convert once to bf16 (~134 MB, fully
// L2-resident in 192 MB L2); logits store (823 MB) is the only big HBM term.
// GEMMs: bf16 WMMA (v_wmma_f32_16x16x32_bf16) with LDS-staged, double-buffered
// tiles fed by global_load_async_to_lds_b128 (ASYNCcnt + s_wait_asynccnt).
// Recurrence: 3-pass chunked scan (16x parallelism over the sequence).
// ---------------------------------------------------------------------------

typedef __bf16 bf16_t;
typedef __attribute__((ext_vector_type(16))) __bf16 v16bf;
typedef __attribute__((ext_vector_type(8)))  __bf16 v8bf;
typedef __attribute__((ext_vector_type(8)))  float  v8f;
typedef int v4i __attribute__((vector_size(4 * sizeof(int))));

#define HID    768
#define VOCAB  50257
#define VOCABP 50304      // multiple of 128
#define LAYERS 12
#define BATCH  2
#define SEQ    2048
#define MTOT   (BATCH*SEQ)   // 4096
#define HH     ((size_t)HID*HID)
#define NCH    16
#define CHLEN  (SEQ/NCH)     // 128

// LDS tile geometry: 128 rows x 32 K-elems, padded row pitch 40 elems (80 B)
// -> the 16B fragment reads (stride 80 B across lanes) are bank-conflict-free.
#define TROWS  128
#define TPITCH 40

#if __has_builtin(__builtin_amdgcn_global_load_async_to_lds_b128) && \
    __has_builtin(__builtin_amdgcn_s_wait_asynccnt)
#define HAS_ASYNC_LDS 1
#endif

typedef __attribute__((address_space(1))) v4i* gptr_v4i;
typedef __attribute__((address_space(3))) v4i* lptr_v4i;

union FragU { v16bf v; v8bf h[2]; };

__device__ __forceinline__ void copy16(bf16_t* lds, const bf16_t* g) {
#ifdef HAS_ASYNC_LDS
  __builtin_amdgcn_global_load_async_to_lds_b128((gptr_v4i)g, (lptr_v4i)lds,
                                                 0, 0);
#else
  *(v8bf*)lds = *(const v8bf*)g;
#endif
}

template <int N>
__device__ __forceinline__ void wait_copies() {
#ifdef HAS_ASYNC_LDS
  __builtin_amdgcn_s_wait_asynccnt(N);
#endif
}

// Read a 16x32 bf16 fragment from an LDS tile (ISA 7.12.2 lane layout):
// lane l<16: row=base+l, K {0..7,16..23}; lane l>=16: K {8..15,24..31}.
__device__ __forceinline__ v16bf lds_frag(const bf16_t* tile, int row, int lane) {
  int r = lane & 15, half = lane >> 4;
  const bf16_t* p = tile + (row + r) * TPITCH + half * 8;
  FragU f;
  f.h[0] = *(const v8bf*)(p);
  f.h[1] = *(const v8bf*)(p + 16);
  return f.v;
}

__device__ __forceinline__ v8f wmma_bf16(v16bf a, v16bf b, v8f c) {
  return __builtin_amdgcn_wmma_f32_16x16x32_bf16(false, a, false, b,
                                                 (short)0, c, false, false);
}

__device__ __forceinline__ float sigmoidf_(float x) {
  return 1.0f / (1.0f + __expf(-x));
}

// ---------------------------------------------------------------------------
// Unified WMMA GEMM:  C(M x N) = A(M x 768 bf16) x Bt(N x 768 bf16)^T
// Block tile 128(M) x 128(N), 8 waves each computing 64(M) x 32(N) = 8 WMMAs
// per 32-wide K step. A/B tiles double-buffered in LDS via async copies.
// MODE 0: store f32 (ld 768)   MODE 1: store bf16 (ld 768)
// MODE 2: head: f32 + bias, col < VOCAB, ld VOCAB
// ---------------------------------------------------------------------------
template <int MODE>
__global__ __launch_bounds__(256)
void k_gemm(const bf16_t* __restrict__ A, const bf16_t* __restrict__ Bt,
            const float* __restrict__ bias, float* __restrict__ outF,
            bf16_t* __restrict__ outB) {
  __shared__ bf16_t smem[2][2][TROWS * TPITCH];   // [buf][A/B][...]  40 KB

  const int tid  = threadIdx.x;
  const int lane = tid & 31, wave = tid >> 5;
  const int r = lane & 15, half = lane >> 4;
  const int wm = (wave & 1) * 64;        // 2 waves cover 128 M
  const int wn = (wave >> 1) * 32;       // 4 waves cover 128 N
  const int aRow0 = blockIdx.y * 128;
  const int bRow0 = blockIdx.x * 128;

  // stage one 128x32 A tile + 128x32 B tile: 512 16B chunks each,
  // 2 chunks per thread per tile -> 4 copy instructions / thread / K-step.
  auto stage = [&](int buf, int kk) {
#pragma unroll
    for (int u = 0; u < 2; ++u) {
      int c = tid * 2 + u;
      int row = c >> 2, sub = (c & 3) * 8;
      copy16(&smem[buf][0][row * TPITCH + sub],
             A + (size_t)(aRow0 + row) * HID + kk + sub);
      copy16(&smem[buf][1][row * TPITCH + sub],
             Bt + (size_t)(bRow0 + row) * HID + kk + sub);
    }
  };

  v8f acc[4][2] = {};

  stage(0, 0);
  const int KSTEPS = HID / 32;   // 24
  for (int s = 0; s < KSTEPS; ++s) {
    if (s + 1 < KSTEPS) {
      stage((s + 1) & 1, (s + 1) * 32);
      wait_copies<4>();   // the 4 older copies (current buffer) are complete
    } else {
      wait_copies<0>();
    }
    __syncthreads();

    const bf16_t* sA = smem[s & 1][0];
    const bf16_t* sB = smem[s & 1][1];
    v16bf bfr[2];
#pragma unroll
    for (int bn = 0; bn < 2; ++bn) bfr[bn] = lds_frag(sB, wn + bn * 16, lane);
#pragma unroll
    for (int am = 0; am < 4; ++am) {
      v16bf afr = lds_frag(sA, wm + am * 16, lane);
      acc[am][0] = wmma_bf16(afr, bfr[0], acc[am][0]);
      acc[am][1] = wmma_bf16(afr, bfr[1], acc[am][1]);
    }
    __syncthreads();   // all reads of this buffer done before it is re-staged
  }

  // epilogue: c[v] -> row = base + half*8 + v, col = base + (lane&15)
#pragma unroll
  for (int am = 0; am < 4; ++am) {
#pragma unroll
    for (int bn = 0; bn < 2; ++bn) {
#pragma unroll
      for (int v = 0; v < 8; ++v) {
        int row = aRow0 + wm + am * 16 + half * 8 + v;
        int col = bRow0 + wn + bn * 16 + r;
        float val = acc[am][bn][v];
        if (MODE == 0) {
          outF[(size_t)row * HID + col] = val;
        } else if (MODE == 1) {
          outB[(size_t)row * HID + col] = (bf16_t)val;
        } else {
          if (col < VOCAB)
            outF[(size_t)row * VOCAB + col] = val + bias[col];
        }
      }
    }
  }
}

// ---------------------------------------------------------------------------
// Tiled transpose + f32->bf16:  out[c*R + r] = (c < C) ? in[r*C + c] : 0
// ---------------------------------------------------------------------------
__global__ __launch_bounds__(256)
void k_transpose_bf16(const float* __restrict__ in, bf16_t* __restrict__ out,
                      int R, int C, int Cp) {
  __shared__ float tile[32][33];
  int r0 = blockIdx.x * 32, c0 = blockIdx.y * 32;
#pragma unroll
  for (int j = 0; j < 32; j += 8) {
    int rr = r0 + threadIdx.y + j, cc = c0 + threadIdx.x;
    tile[threadIdx.y + j][threadIdx.x] =
        (rr < R && cc < C) ? in[(size_t)rr * C + cc] : 0.0f;
  }
  __syncthreads();
#pragma unroll
  for (int j = 0; j < 32; j += 8) {
    int cc = c0 + threadIdx.y + j, rr = r0 + threadIdx.x;
    if (cc < Cp && rr < R)
      out[(size_t)cc * R + rr] = (bf16_t)tile[threadIdx.x][threadIdx.y + j];
  }
}

// ---------------------------------------------------------------------------
__global__ __launch_bounds__(256)
void k_embed(const int* __restrict__ ids, const float* __restrict__ emb,
             bf16_t* __restrict__ xb) {
  int tid = blockIdx.x * 256 + threadIdx.x;
  if (tid >= MTOT * HID) return;
  int tok = tid / HID, h = tid - tok * HID;
  xb[tid] = (bf16_t)emb[(size_t)ids[tok] * HID + h];
}

// ---------------------------------------------------------------------------
// Chunked parallel scan over the recurrence  h_t = sig(f)*h + silu(i)*(1-sig(f))
// pass1: per (b,chunk,h): local scan from 0, record (P = prod fs, Hend)
// pass2: per (b,h): serial combine over 16 chunks -> per-chunk h0
// pass3: per (b,chunk,h): rescan with true h0, emit bf16(g * h)
// ---------------------------------------------------------------------------
__global__ __launch_bounds__(256)
void k_scan1(const float* __restrict__ F, const float* __restrict__ I,
             float* __restrict__ Pc, float* __restrict__ Hc) {
  int idx = blockIdx.x * 256 + threadIdx.x;            // (b*NCH + ch)*HID + h
  if (idx >= BATCH * NCH * HID) return;
  int h = idx % HID, bc = idx / HID;
  int ch = bc % NCH, b = bc / NCH;
  size_t base = ((size_t)b * SEQ + (size_t)ch * CHLEN) * HID + h;
  float P = 1.0f, hl = 0.0f;
  for (int t = 0; t < CHLEN; ++t) {
    size_t p = base + (size_t)t * HID;
    float fs = sigmoidf_(F[p]);
    float xi = I[p];
    float im = xi * sigmoidf_(xi) * (1.0f - fs);
    hl = fs * hl + im;
    P *= fs;
  }
  Pc[idx] = P;
  Hc[idx] = hl;
}

__global__ __launch_bounds__(256)
void k_scan2(const float* __restrict__ Pc, const float* __restrict__ Hc,
             float* __restrict__ H0c) {
  int idx = blockIdx.x * 256 + threadIdx.x;            // b*HID + h
  if (idx >= BATCH * HID) return;
  int h = idx % HID, b = idx / HID;
  float h0 = 0.0f;
  for (int ch = 0; ch < NCH; ++ch) {
    size_t i = ((size_t)b * NCH + ch) * HID + h;
    H0c[i] = h0;
    h0 = Pc[i] * h0 + Hc[i];
  }
}

__global__ __launch_bounds__(256)
void k_scan3(const float* __restrict__ F, const float* __restrict__ I,
             const float* __restrict__ G, const float* __restrict__ H0c,
             bf16_t* __restrict__ Ab) {
  int idx = blockIdx.x * 256 + threadIdx.x;
  if (idx >= BATCH * NCH * HID) return;
  int h = idx % HID, bc = idx / HID;
  int ch = bc % NCH, b = bc / NCH;
  size_t base = ((size_t)b * SEQ + (size_t)ch * CHLEN) * HID + h;
  float hc = H0c[idx];
  for (int t = 0; t < CHLEN; ++t) {
    size_t p = base + (size_t)t * HID;
    float fs = sigmoidf_(F[p]);
    float xi = I[p];
    float im = xi * sigmoidf_(xi) * (1.0f - fs);
    hc = fs * hc + im;
    Ab[p] = (bf16_t)(G[p] * hc);
  }
}

// ---------------------------------------------------------------------------
extern "C" void kernel_launch(void* const* d_in, const int* in_sizes, int n_in,
                              void* d_out, int out_size, void* d_ws, size_t ws_size,
                              hipStream_t stream) {
  const int*   ids   = (const int*)d_in[0];
  const float* emb   = (const float*)d_in[1];
  const float* Wi    = (const float*)d_in[2];
  const float* Wf    = (const float*)d_in[3];
  const float* Wg    = (const float*)d_in[4];
  const float* Wo    = (const float*)d_in[5];
  const float* Whead = (const float*)d_in[6];
  const float* bhead = (const float*)d_in[7];
  float* logits = (float*)d_out;

  char* ws = (char*)d_ws;
  size_t off = 0;
  auto carve = [&](size_t bytes) {
    void* p = ws + off;
    off = (off + bytes + 255) & ~(size_t)255;
    return p;
  };
  bf16_t* WtAll  = (bf16_t*)carve(48 * HH * sizeof(bf16_t));
  bf16_t* WheadT = (bf16_t*)carve((size_t)VOCABP * HID * sizeof(bf16_t));
  bf16_t* Xbf    = (bf16_t*)carve((size_t)MTOT * HID * sizeof(bf16_t));
  bf16_t* Abf    = (bf16_t*)carve((size_t)MTOT * HID * sizeof(bf16_t));
  float*  Ibuf   = (float*)carve((size_t)MTOT * HID * sizeof(float));
  float*  Fbuf   = (float*)carve((size_t)MTOT * HID * sizeof(float));
  float*  Gbuf   = (float*)carve((size_t)MTOT * HID * sizeof(float));
  float*  Pc     = (float*)carve((size_t)BATCH * NCH * HID * sizeof(float));
  float*  Hc     = (float*)carve((size_t)BATCH * NCH * HID * sizeof(float));
  float*  H0c    = (float*)carve((size_t)BATCH * NCH * HID * sizeof(float));

  dim3 tb(32, 8);
  for (int l = 0; l < LAYERS; ++l) {
    const float* src[4] = {Wi + l * HH, Wf + l * HH, Wg + l * HH, Wo + l * HH};
    for (int m = 0; m < 4; ++m)
      k_transpose_bf16<<<dim3(HID / 32, HID / 32), tb, 0, stream>>>(
          src[m], WtAll + (size_t)(l * 4 + m) * HH, HID, HID, HID);
  }
  k_transpose_bf16<<<dim3(HID / 32, VOCABP / 32), tb, 0, stream>>>(
      Whead, WheadT, HID, VOCAB, VOCABP);

  k_embed<<<(MTOT * HID + 255) / 256, 256, 0, stream>>>(ids, emb, Xbf);

  const dim3 gH(HID / 128, MTOT / 128);        // 6 x 32
  const int scanBlocks = (BATCH * NCH * HID + 255) / 256;   // 96
  for (int l = 0; l < LAYERS; ++l) {
    const bf16_t* Bti = WtAll + (size_t)(l * 4 + 0) * HH;
    const bf16_t* Btf = WtAll + (size_t)(l * 4 + 1) * HH;
    const bf16_t* Btg = WtAll + (size_t)(l * 4 + 2) * HH;
    const bf16_t* Bto = WtAll + (size_t)(l * 4 + 3) * HH;
    k_gemm<0><<<gH, 256, 0, stream>>>(Xbf, Bti, nullptr, Ibuf, nullptr);
    k_gemm<0><<<gH, 256, 0, stream>>>(Xbf, Btf, nullptr, Fbuf, nullptr);
    k_gemm<0><<<gH, 256, 0, stream>>>(Xbf, Btg, nullptr, Gbuf, nullptr);
    k_scan1<<<scanBlocks, 256, 0, stream>>>(Fbuf, Ibuf, Pc, Hc);
    k_scan2<<<(BATCH * HID + 255) / 256, 256, 0, stream>>>(Pc, Hc, H0c);
    k_scan3<<<scanBlocks, 256, 0, stream>>>(Fbuf, Ibuf, Gbuf, H0c, Abf);
    k_gemm<1><<<gH, 256, 0, stream>>>(Abf, Bto, nullptr, nullptr, Xbf);
  }

  k_gemm<2><<<dim3(VOCABP / 128, MTOT / 128), 256, 0, stream>>>(
      Xbf, WheadT, bhead, logits, nullptr);
}
